// View_selector_37477884625438
// MI455X (gfx1250) — matrix operand order repaired and checked by reading, not compiled, not run on hardware
//
#include <hip/hip_runtime.h>

// ---------------------------------------------------------------------------
// View selector pipeline for MI455X (gfx1250, wave32, WMMA + async-to-LDS).
// GEMMs use v_wmma_f32_16x16x32_bf16; bf16 operand staging uses
// global_load_async_to_lds_b128 (ASYNCcnt) where the source is already bf16.
// ---------------------------------------------------------------------------

#define S_VIEWS   10
#define NCLASSES  40
#define FEAT      512
#define BATCH     2048
#define NPTS      40
#define KNN_K     4
#define M_ROWS    (BATCH * KNN_K)      // 8192
#define K1        (S_VIEWS * FEAT)     // 5120
#define N1        (256 * S_VIEWS)      // 2560
#define N2        (NCLASSES * S_VIEWS) // 400
#define N2P       448                  // padded to 7*64: no load guards in GEMM2

typedef __attribute__((ext_vector_type(16))) __bf16 v16bf;
typedef __attribute__((ext_vector_type(8)))  __bf16 v8bf;
typedef __attribute__((ext_vector_type(8)))  float  v8f;

static __device__ __forceinline__ v16bf cat8(v8bf lo, v8bf hi) {
  return __builtin_shufflevector(lo, hi, 0,1,2,3,4,5,6,7,8,9,10,11,12,13,14,15);
}

// CDNA5 async copy global -> LDS (16B per lane), tracked by ASYNCcnt.
// Low 32 bits of a flat pointer into LDS are the wave-relative LDS offset
// (ISA 10.2: LDS aperture maps addr[31:0] directly).
static __device__ __forceinline__ void async_copy_b128(void* lds_dst, const void* gsrc) {
  unsigned loff = (unsigned)(size_t)lds_dst;
  asm volatile("global_load_async_to_lds_b128 %0, %1, off"
               :: "v"(loff), "v"(gsrc) : "memory");
}
static __device__ __forceinline__ void wait_async_0(void) {
  asm volatile("s_wait_asynccnt 0x0" ::: "memory");
}
static __device__ __forceinline__ void wait_async_le3(void) {
  asm volatile("s_wait_asynccnt 0x3" ::: "memory");
}

// ---------------------------------------------------------------------------
// Kernel 1: deterministic FPS (start at index 0) + 4-NN with JAX tie-breaking.
// One thread per batch (N=40 is tiny).
// ---------------------------------------------------------------------------
__global__ __launch_bounds__(256)
void fps_knn_kernel(const float* __restrict__ verts, int* __restrict__ knn_out) {
  int b = blockIdx.x * blockDim.x + threadIdx.x;
  if (b >= BATCH) return;
  const float* vb = verts + (size_t)b * NPTS * 3;
  float px[NPTS], py[NPTS], pz[NPTS], dist[NPTS];
  for (int i = 0; i < NPTS; ++i) {
    px[i] = vb[3*i]; py[i] = vb[3*i+1]; pz[i] = vb[3*i+2];
    dist[i] = 1e10f;
  }
  int cent[S_VIEWS];
  int far = 0;
  for (int it = 0; it < S_VIEWS; ++it) {
    cent[it] = far;
    float cx = px[far], cy = py[far], cz = pz[far];
    float best = -1.0f; int bi = 0;
    for (int n = 0; n < NPTS; ++n) {
      float dx = px[n]-cx, dy = py[n]-cy, dz = pz[n]-cz;
      float d = dx*dx + dy*dy + dz*dz;
      float dn = fminf(dist[n], d);
      dist[n] = dn;
      if (dn > best) { best = dn; bi = n; }   // strict > == argmax first occurrence
    }
    far = bi;
  }
  for (int v = 0; v < S_VIEWS; ++v) {
    int c = cent[v];
    float cx = px[c], cy = py[c], cz = pz[c];
    float bd[KNN_K] = {1e30f, 1e30f, 1e30f, 1e30f};
    int   bi[KNN_K] = {0, 0, 0, 0};
    for (int n = 0; n < NPTS; ++n) {
      float dx = px[n]-cx, dy = py[n]-cy, dz = pz[n]-cz;
      float d = dx*dx + dy*dy + dz*dz;
      if (d < bd[KNN_K-1]) {                  // strict < keeps earlier index on tie
        int p = KNN_K - 1;
        while (p > 0 && d < bd[p-1]) { bd[p] = bd[p-1]; bi[p] = bi[p-1]; --p; }
        bd[p] = d; bi[p] = n;
      }
    }
    for (int j = 0; j < KNN_K; ++j) knn_out[((size_t)b * S_VIEWS + v) * KNN_K + j] = bi[j];
  }
}

// ---------------------------------------------------------------------------
// Kernel 2a: W1 f32 [K1][N1] -> bf16 transposed W1T [N1][K1] (LDS-tiled).
// ---------------------------------------------------------------------------
__global__ __launch_bounds__(256)
void transpose_w1_kernel(const float* __restrict__ W1, __bf16* __restrict__ W1T) {
  __shared__ float t[32][33];
  const int n0 = blockIdx.x * 32;          // over N1
  const int k0 = blockIdx.y * 32;          // over K1
  const int tx = threadIdx.x & 31;
  const int ty = threadIdx.x >> 5;         // 8 rows per pass
#pragma unroll
  for (int r = 0; r < 32; r += 8)
    t[ty + r][tx] = W1[(size_t)(k0 + ty + r) * N1 + n0 + tx];
  __syncthreads();
#pragma unroll
  for (int r = 0; r < 32; r += 8)
    W1T[(size_t)(n0 + ty + r) * K1 + k0 + tx] = (__bf16)t[tx][ty + r];
}

// ---------------------------------------------------------------------------
// Kernel 2b: W2 f32 [N1][N2] -> bf16 transposed + padded W2T [N2P][N1].
// ---------------------------------------------------------------------------
__global__ __launch_bounds__(256)
void transpose_w2_kernel(const float* __restrict__ W2, __bf16* __restrict__ W2T) {
  __shared__ float t[32][33];
  const int n0 = blockIdx.x * 32;          // over N2P
  const int k0 = blockIdx.y * 32;          // over N1 (K of GEMM2)
  const int tx = threadIdx.x & 31;
  const int ty = threadIdx.x >> 5;
#pragma unroll
  for (int r = 0; r < 32; r += 8)
    t[ty + r][tx] = (n0 + tx < N2) ? W2[(size_t)(k0 + ty + r) * N2 + n0 + tx] : 0.0f;
  __syncthreads();
#pragma unroll
  for (int r = 0; r < 32; r += 8)
    W2T[(size_t)(n0 + ty + r) * N1 + k0 + tx] = (__bf16)t[tx][ty + r];
}

// ---------------------------------------------------------------------------
// Kernel 3: GEMM1 — h = leaky_relu(F1 @ W1 + b1), stored bf16.
// M=8192, K=5120, N=2560. Block tile 128x128, 8 waves (2x4), wave = 64x32.
// A-tile: gathered from F via knn, f32->bf16 via VALU (view uniform per K-step).
// B-tile: 2 async b128 copies per thread from pre-transposed W1T.
// ---------------------------------------------------------------------------
__global__ __launch_bounds__(256)
void gemm1_kernel(const float* __restrict__ F, const int* __restrict__ knn,
                  const __bf16* __restrict__ W1T, const float* __restrict__ b1,
                  __bf16* __restrict__ Hbf) {
  __shared__ __bf16 sA[128 * 40];   // [m][k], stride 40 halves for banking
  __shared__ __bf16 sB[128 * 40];   // [n][k]
  const int tid  = threadIdx.x;
  const int lane = tid & 31;
  const int wave = tid >> 5;
  const int wm = wave >> 2;              // 0..1
  const int wn = wave & 3;               // 0..3
  const int lh = (lane >> 4) & 1;
  const int lm = lane & 15;
  const int mb = blockIdx.y * 128;
  const int nb = blockIdx.x * 128;

  int growR[2], kbAR[2], mAR[2], nBR[2], kbBR[2];
#pragma unroll
  for (int r = 0; r < 2; ++r) {
    int cid = r * 256 + tid;             // A chunks of 8 halves
    mAR[r]  = cid >> 2;
    kbAR[r] = (cid & 3) * 8;
    growR[r] = mb + mAR[r];
    nBR[r]  = cid & 127;                 // B chunks of 8 halves
    kbBR[r] = (cid >> 7) * 8;
  }

  v8f acc[4][2] = {};
  int curV = -1;
  int ptR[2] = {0, 0};

  for (int kt = 0; kt < K1 / 32; ++kt) {
    const int k0 = kt * 32;
    const int vview = k0 >> 9;           // uniform view index for this K-step
    const int cbase = k0 & 511;
    // B tile first: async copies fly while we do the A gather/convert.
#pragma unroll
    for (int r = 0; r < 2; ++r)
      async_copy_b128(sB + nBR[r] * 40 + kbBR[r],
                      W1T + (size_t)(nb + nBR[r]) * K1 + k0 + kbBR[r]);
    if (vview != curV) {                 // uniform branch: refresh gather indices
      curV = vview;
#pragma unroll
      for (int r = 0; r < 2; ++r) {
        int g = growR[r];
        ptR[r] = knn[(((g >> 2) * S_VIEWS) + vview) * KNN_K + (g & 3)];
      }
    }
    // A tile: gather rows of F, convert f32->bf16.
#pragma unroll
    for (int r = 0; r < 2; ++r) {
      int g = growR[r];
      const float* src = F + (size_t)(((g >> 2) * NPTS + ptR[r]) * FEAT + cbase + kbAR[r]);
      float4 f0 = *(const float4*)src;
      float4 f1 = *(const float4*)(src + 4);
      if (kt + 1 < K1 / 32) __builtin_prefetch(src + 32, 0, 2);  // global_prefetch_b8
      v8bf pk;
      pk[0] = (__bf16)f0.x; pk[1] = (__bf16)f0.y; pk[2] = (__bf16)f0.z; pk[3] = (__bf16)f0.w;
      pk[4] = (__bf16)f1.x; pk[5] = (__bf16)f1.y; pk[6] = (__bf16)f1.z; pk[7] = (__bf16)f1.w;
      *(v8bf*)(sA + mAR[r] * 40 + kbAR[r]) = pk;
    }
    wait_async_0();                      // our B chunks landed in LDS
    __syncthreads();                     // + everyone's A stores (dscnt)

    v16bf afrag[4], bfrag[2];
#pragma unroll
    for (int mt = 0; mt < 4; ++mt) {
      const __bf16* p = sA + (wm * 64 + mt * 16 + lm) * 40 + lh * 8;   // A: K{0..7|8..15},{+16}
      afrag[mt] = cat8(*(const v8bf*)p, *(const v8bf*)(p + 16));
    }
#pragma unroll
    for (int nt = 0; nt < 2; ++nt) {
      const __bf16* p = sB + (wn * 32 + nt * 16 + lm) * 40 + lh * 8;   // B: K{0..7|8..15},{+16}
      bfrag[nt] = cat8(*(const v8bf*)p, *(const v8bf*)(p + 16));
    }
#pragma unroll
    for (int mt = 0; mt < 4; ++mt)
#pragma unroll
      for (int nt = 0; nt < 2; ++nt)
        acc[mt][nt] = __builtin_amdgcn_wmma_f32_16x16x32_bf16(
            false, afrag[mt], false, bfrag[nt], (short)0, acc[mt][nt], false, false);
    __syncthreads();
  }

  // Epilogue: +bias, leaky_relu(0.2), store bf16.
#pragma unroll
  for (int mt = 0; mt < 4; ++mt) {
#pragma unroll
    for (int nt = 0; nt < 2; ++nt) {
      int col = nb + wn * 32 + nt * 16 + lm;
      float bias = b1[col];
#pragma unroll
      for (int r = 0; r < 8; ++r) {
        int row = mb + wm * 64 + mt * 16 + lh * 8 + r;
        float x = acc[mt][nt][r] + bias;
        x = (x >= 0.0f) ? x : 0.2f * x;
        Hbf[(size_t)row * N1 + col] = (__bf16)x;
      }
    }
  }
}

// ---------------------------------------------------------------------------
// Kernel 4: GEMM2 — F_score = h @ W2 + b2, permuted into d_out.
// M=8192, K=2560, N=448(padded). Block 128x64, 8 waves (4x2), wave 32x32.
// Fully async, double-buffered LDS: 3 async b128 per thread per tile,
// in-order completion => s_wait_asynccnt 3 retires the previous tile.
// ---------------------------------------------------------------------------
__global__ __launch_bounds__(256)
void gemm2_kernel(const __bf16* __restrict__ Hbf, const __bf16* __restrict__ W2T,
                  const float* __restrict__ b2, float* __restrict__ Fscore) {
  __shared__ __bf16 sA[2][128 * 40];
  __shared__ __bf16 sB[2][64 * 40];
  const int tid  = threadIdx.x;
  const int lane = tid & 31;
  const int wave = tid >> 5;
  const int wm = wave >> 1;              // 0..3
  const int wn = wave & 1;               // 0..1
  const int lh = (lane >> 4) & 1;
  const int lm = lane & 15;
  const int mb = blockIdx.y * 128;
  const int nb = blockIdx.x * 64;
  const int nB  = tid & 63;
  const int kbB = (tid >> 6) * 8;
  const int KT  = N1 / 32;               // 80

  auto issue_tile = [&](int kt, int buf) {
    const int k0 = kt * 32;
#pragma unroll
    for (int r = 0; r < 2; ++r) {
      int cid = r * 256 + tid;
      int m = cid >> 2, kb = (cid & 3) * 8;
      async_copy_b128(&sA[buf][m * 40 + kb], Hbf + (size_t)(mb + m) * N1 + k0 + kb);
    }
    async_copy_b128(&sB[buf][nB * 40 + kbB], W2T + (size_t)(nb + nB) * N1 + k0 + kbB);
  };

  v8f acc[2][2] = {};
  issue_tile(0, 0);
  for (int kt = 0; kt < KT; ++kt) {
    const int cur = kt & 1;
    const int nk = (kt + 1 < KT) ? kt + 1 : kt;   // last iter: harmless re-issue
    issue_tile(nk, cur ^ 1);
    wait_async_le3();                    // previous 3 (tile kt) complete, in order
    __syncthreads();

    v16bf afrag[2], bfrag[2];
#pragma unroll
    for (int mt = 0; mt < 2; ++mt) {
      const __bf16* p = &sA[cur][(wm * 32 + mt * 16 + lm) * 40 + lh * 8];
      afrag[mt] = cat8(*(const v8bf*)p, *(const v8bf*)(p + 16));
    }
#pragma unroll
    for (int nt = 0; nt < 2; ++nt) {
      const __bf16* p = &sB[cur][(wn * 32 + nt * 16 + lm) * 40 + lh * 8];
      bfrag[nt] = cat8(*(const v8bf*)p, *(const v8bf*)(p + 16));
    }
#pragma unroll
    for (int mt = 0; mt < 2; ++mt)
#pragma unroll
      for (int nt = 0; nt < 2; ++nt)
        acc[mt][nt] = __builtin_amdgcn_wmma_f32_16x16x32_bf16(
            false, afrag[mt], false, bfrag[nt], (short)0, acc[mt][nt], false, false);
    __syncthreads();                     // readers done before buf reuse
  }
  wait_async_0();                        // drain the final redundant issue

  // Epilogue: +b2, permute (b,k,s,nc)->(b,s,k,nc), drop padded columns.
#pragma unroll
  for (int mt = 0; mt < 2; ++mt) {
#pragma unroll
    for (int nt = 0; nt < 2; ++nt) {
      int col = nb + wn * 32 + nt * 16 + lm;
      if (col < N2) {
        float bias = b2[col];
        int v = col / NCLASSES, c = col % NCLASSES;
#pragma unroll
        for (int r = 0; r < 8; ++r) {
          int row = mb + wm * 32 + mt * 16 + lh * 8 + r;
          int bb = row >> 2, j = row & 3;
          Fscore[((size_t)bb * S_VIEWS + v) * (KNN_K * NCLASSES) + j * NCLASSES + c] =
              acc[mt][nt][r] + bias;
        }
      }
    }
  }
}

// ---------------------------------------------------------------------------
// Kernel 5: softmax over views -> max over classes -> argmax over k -> gather.
// One block per batch.
// ---------------------------------------------------------------------------
__global__ __launch_bounds__(256)
void select_kernel(const float* __restrict__ F, const float* __restrict__ verts,
                   const float* __restrict__ Fscore, const int* __restrict__ knn,
                   float* __restrict__ Fnew, float* __restrict__ Vnew) {
  const int b = blockIdx.x;
  const int t = threadIdx.x;
  __shared__ float s[S_VIEWS * KNN_K * NCLASSES];   // [v][j][c]
  __shared__ float f1v[S_VIEWS * KNN_K];
  __shared__ int   sel[S_VIEWS];

  const float* fs = Fscore + (size_t)b * (S_VIEWS * KNN_K * NCLASSES);
  for (int i = t; i < S_VIEWS * KNN_K * NCLASSES; i += 256) s[i] = fs[i];
  __syncthreads();

  if (t < KNN_K * NCLASSES) {            // t == j*40 + c; softmax over v
    float m = -3.402823466e38f;
    for (int v = 0; v < S_VIEWS; ++v) m = fmaxf(m, s[v * 160 + t]);
    float e[S_VIEWS]; float den = 0.0f;
    for (int v = 0; v < S_VIEWS; ++v) { e[v] = expf(s[v * 160 + t] - m); den += e[v]; }
    for (int v = 0; v < S_VIEWS; ++v) s[v * 160 + t] = e[v] / den;
  }
  __syncthreads();
  if (t < S_VIEWS * KNN_K) {             // max over classes
    int v = t >> 2, j = t & 3;
    float mx = -3.402823466e38f;
    for (int c = 0; c < NCLASSES; ++c) mx = fmaxf(mx, s[v * 160 + j * 40 + c]);
    f1v[t] = mx;
  }
  __syncthreads();
  if (t < S_VIEWS) {                     // argmax over k (first occurrence)
    float best = f1v[t * 4]; int bj = 0;
    for (int j = 1; j < KNN_K; ++j)
      if (f1v[t * 4 + j] > best) { best = f1v[t * 4 + j]; bj = j; }
    sel[t] = knn[((size_t)b * S_VIEWS + t) * KNN_K + bj];
  }
  __syncthreads();

  for (int e = t; e < S_VIEWS * FEAT; e += 256) {
    int v = e >> 9, c = e & (FEAT - 1);
    Fnew[(size_t)b * (S_VIEWS * FEAT) + e] = F[((size_t)b * NPTS + sel[v]) * FEAT + c];
  }
  if (t < S_VIEWS * 3) {
    int v = t / 3, c = t % 3;
    Vnew[(size_t)b * (S_VIEWS * 3) + t] = verts[((size_t)b * NPTS + sel[v]) * 3 + c];
  }
}

// ---------------------------------------------------------------------------
// Launcher. Workspace layout (bytes, total ~71 MB):
//   [0,          327680)   knn indices (int, B*10*4)
//   [524288,   42467328)   h bf16        (8192*2560*2)
//   [42467328, 44761088)   W2T bf16 pad  (448*2560*2)
//   [44761088, 70975488)   W1T bf16      (2560*5120*2)
// d_out: F_new [0,10485760) | F_score [10485760,13762560) | V_new [...,13824000)
// ---------------------------------------------------------------------------
extern "C" void kernel_launch(void* const* d_in, const int* in_sizes, int n_in,
                              void* d_out, int out_size, void* d_ws, size_t ws_size,
                              hipStream_t stream) {
  (void)in_sizes; (void)n_in; (void)out_size; (void)ws_size;
  const float* F  = (const float*)d_in[0];
  const float* V  = (const float*)d_in[1];
  const float* W1 = (const float*)d_in[2];
  const float* b1 = (const float*)d_in[3];
  const float* W2 = (const float*)d_in[4];
  const float* b2 = (const float*)d_in[5];

  char* ws = (char*)d_ws;
  const size_t HB_OFF = (size_t)512 * 1024;
  const size_t W2_OFF = HB_OFF + (size_t)M_ROWS * N1 * 2;
  const size_t W1_OFF = W2_OFF + (size_t)N2P * N1 * 2;
  int*    knn  = (int*)ws;
  __bf16* Hbf  = (__bf16*)(ws + HB_OFF);
  __bf16* W2T  = (__bf16*)(ws + W2_OFF);
  __bf16* W1T  = (__bf16*)(ws + W1_OFF);

  float* Fnew   = (float*)d_out;
  float* Fscore = (float*)d_out + (size_t)BATCH * S_VIEWS * FEAT;
  float* Vnew   = Fscore + (size_t)BATCH * S_VIEWS * KNN_K * NCLASSES;

  fps_knn_kernel<<<BATCH / 256, 256, 0, stream>>>(V, knn);
  transpose_w1_kernel<<<dim3(N1 / 32, K1 / 32), 256, 0, stream>>>(W1, W1T);
  transpose_w2_kernel<<<dim3(N2P / 32, N1 / 32), 256, 0, stream>>>(W2, W2T);
  gemm1_kernel<<<dim3(N1 / 128, M_ROWS / 128), 256, 0, stream>>>(F, knn, W1T, b1, Hbf);
  gemm2_kernel<<<dim3(N2P / 64, M_ROWS / 128), 256, 0, stream>>>(Hbf, W2T, b2, Fscore);
  select_kernel<<<BATCH, 256, 0, stream>>>(F, V, Fscore, knn, Fnew, Vnew);
}
